// Sequence_52561809768508
// MI455X (gfx1250) — matrix operand order repaired and checked by reading, not compiled, not run on hardware
//
#include <hip/hip_runtime.h>
#include <hip/hip_bf16.h>

typedef __attribute__((ext_vector_type(16))) _Float16 v16h;
typedef __attribute__((ext_vector_type(8)))  float    v8f;
typedef __attribute__((ext_vector_type(4)))  int      v4i;

#define TT   512   // timesteps
#define DD   64    // input feature dim
#define HH   70    // hidden size
#define GP   320   // 4*H=280 padded to 320 -> 20 N-tiles, 5 per wave (8 waves)
#define KP   160   // concat-K (64+70 or 70+70) padded to 32 multiple
#define BT   32    // batch tile per workgroup
#define NTHR 256   // 8 waves

#if __has_builtin(__builtin_amdgcn_global_load_async_to_lds_b128)
#define HAVE_ASYNC_LDS 1
#else
#define HAVE_ASYNC_LDS 0
#endif

// Address-space-qualified vector pointee types for the async-LDS builtin,
// whose signature is (v4i AS1*, v4i AS3*, imm offset, imm cpol).
typedef __attribute__((address_space(1))) v4i v4i_glob;
typedef __attribute__((address_space(3))) v4i v4i_lds;

// ---------------------------------------------------------------------------
// Prep: pack [W_ih ; W_hh] into padded f16 [GP x KP] matrices + summed biases.
// Layer0 K layout: [0,64)=x, [64,134)=h0, rest 0.
// Layer1 K layout: [0,70)=h0, [70,140)=h1, rest 0.
// ---------------------------------------------------------------------------
__global__ void pack_weights_kernel(const float* __restrict__ Wih0, const float* __restrict__ Whh0,
                                    const float* __restrict__ bih0, const float* __restrict__ bhh0,
                                    const float* __restrict__ Wih1, const float* __restrict__ Whh1,
                                    const float* __restrict__ bih1, const float* __restrict__ bhh1,
                                    _Float16* __restrict__ WA, _Float16* __restrict__ WB,
                                    float* __restrict__ B0, float* __restrict__ B1)
{
    int idx = blockIdx.x * blockDim.x + threadIdx.x;
    if (idx < GP * KP) {
        int g = idx / KP, k = idx % KP;
        float a = 0.f, b = 0.f;
        if (g < 4 * HH) {
            if (k < DD)            a = Wih0[g * DD + k];
            else if (k < DD + HH)  a = Whh0[g * HH + (k - DD)];
            if (k < HH)            b = Wih1[g * HH + k];
            else if (k < 2 * HH)   b = Whh1[g * HH + (k - HH)];
        }
        WA[idx] = (_Float16)a;
        WB[idx] = (_Float16)b;
    }
    if (idx < GP) {
        B0[idx] = (idx < 4 * HH) ? (bih0[idx] + bhh0[idx]) : 0.f;
        B1[idx] = (idx < 4 * HH) ? (bih1[idx] + bhh1[idx]) : 0.f;
    }
}

__device__ __forceinline__ float sigmoidf_(float x) { return 1.f / (1.f + __expf(-x)); }

// Async (ASYNCcnt-tracked) copy of one 16B chunk global -> LDS.
__device__ __forceinline__ void async_cp16(const float* g, float* l) {
#if HAVE_ASYNC_LDS
    __builtin_amdgcn_global_load_async_to_lds_b128(
        (v4i_glob*)g, (v4i_lds*)l, 0, 0);
#else
    *(uint4*)l = *(const uint4*)g;   // synchronous fallback
#endif
}

__device__ __forceinline__ void wait_async_all() {
#if HAVE_ASYNC_LDS
#if __has_builtin(__builtin_amdgcn_s_wait_asynccnt)
    __builtin_amdgcn_s_wait_asynccnt(0);
#else
    asm volatile("s_wait_asynccnt 0" ::: "memory");
#endif
#endif
}

// Prefetch one [BT x DD] f32 x-tile into an LDS staging buffer.
__device__ __forceinline__ void prefetch_x(const float* __restrict__ gsrc,
                                           float* __restrict__ ldst, int tid) {
#pragma unroll
    for (int i = 0; i < (BT * DD * 4) / (16 * NTHR); ++i) {   // 2 chunks/thread
        int chunk = tid + i * NTHR;
        async_cp16(gsrc + chunk * 4, ldst + chunk * 4);
    }
}

union AFrag { v16h h; uint4 q[2]; };

// One wave computes a [16 x 80] slab of gates = act[16 x 160] * wgt^T.
// act: [BT][KP] f16 row-major (activations), wgt: [GP][KP] f16 (gate-major).
__device__ __forceinline__ void gemm_slab(const _Float16* __restrict__ act,
                                          const _Float16* __restrict__ wgt,
                                          float* __restrict__ gates,
                                          int mRow, int nTile0, int half, int l16)
{
    // Hoist the 5 A fragments for this wave's M tile (reused across 5 N tiles).
    AFrag afr[5];
    const _Float16* arow = act + (size_t)(mRow + l16) * KP;
#pragma unroll
    for (int kt = 0; kt < 5; ++kt) {
        int kb = kt * 32;
        // 16-bit A 16x32 layout: lanes 0-15 -> K 0-7 (v0-3) & 16-23 (v4-7);
        // lanes 16-31 -> K 8-15 & 24-31.
        afr[kt].q[0] = *(const uint4*)(arow + kb + half * 8);
        afr[kt].q[1] = *(const uint4*)(arow + kb + 16 + half * 8);
    }
#pragma unroll
    for (int nt = 0; nt < 5; ++nt) {
        int nb = (nTile0 + nt) * 16;
        v8f acc = {0.f, 0.f, 0.f, 0.f, 0.f, 0.f, 0.f, 0.f};
        const _Float16* wrow = wgt + (size_t)(nb + l16) * KP;
#pragma unroll
        for (int kt = 0; kt < 5; ++kt) {
            // 16-bit B 32x16 layout (matches A): lanes 0-15 -> K 0-7 & 16-23,
            // lanes 16-31 -> K 8-15 & 24-31.
            AFrag bfr;
            bfr.q[0] = *(const uint4*)(wrow + kt * 32 + half * 8);
            bfr.q[1] = *(const uint4*)(wrow + kt * 32 + 16 + half * 8);
            acc = __builtin_amdgcn_wmma_f32_16x16x32_f16(
                false, afr[kt].h, false, bfr.h, (short)0, acc, false, false);
        }
        // f32 C/D 16x16 layout: VGPR j -> row j + 8*half, col = lane%16.
#pragma unroll
        for (int j = 0; j < 8; ++j)
            gates[(size_t)(mRow + j + half * 8) * GP + nb + l16] = acc[j];
    }
}

// ---------------------------------------------------------------------------
// Main kernel: each workgroup owns BT batch rows for all TT steps.
// ---------------------------------------------------------------------------
__global__ __launch_bounds__(NTHR, 1)
void lstm_seq_kernel(const float* __restrict__ x,
                     const _Float16* __restrict__ gWA, const _Float16* __restrict__ gWB,
                     const float* __restrict__ gB0, const float* __restrict__ gB1,
                     const float* __restrict__ gW1, const float* __restrict__ gb1,
                     const float* __restrict__ gW2, const float* __restrict__ gb2,
                     float* __restrict__ out, int Bn)
{
    extern __shared__ char smem[];
    _Float16* sWA  = (_Float16*)smem;                // [GP*KP] 102400 B
    _Float16* sWB  = sWA + GP * KP;                  //         102400 B
    _Float16* vec0 = sWB + GP * KP;                  // [BT*KP]  10240 B (x | h0)
    _Float16* vec1 = vec0 + BT * KP;                 //          10240 B (h0 | h1)
    float* gates   = (float*)(vec1 + BT * KP);       // [BT*GP]  40960 B
    float* c0      = gates + BT * GP;                // [BT*HH]   8960 B
    float* c1      = c0 + BT * HH;                   //           8960 B
    float* h1f     = c1 + BT * HH;                   //           8960 B
    float* sB0     = h1f + BT * HH;                  // [GP]      1280 B
    float* sB1     = sB0 + GP;                       //           1280 B
    float* xraw    = sB1 + GP;                       // [2][BT*DD] 16384 B
    // total = 312064 B  (< 320 KB WGP LDS)

    const int tid  = threadIdx.x;
    const int wave = tid >> 5;
    const int lane = tid & 31;
    const int half = lane >> 4;
    const int l16  = lane & 15;
    const int brow = blockIdx.x * BT;

    // Kick off async prefetch of x[0] immediately, overlap with weight staging.
    prefetch_x(x + (size_t)brow * DD, xraw, tid);

    // Stage weights into LDS (one-time).
    {
        const uint4* s = (const uint4*)gWA; uint4* d = (uint4*)sWA;
        for (int i = tid; i < GP * KP / 8; i += NTHR) d[i] = s[i];
        s = (const uint4*)gWB; d = (uint4*)sWB;
        for (int i = tid; i < GP * KP / 8; i += NTHR) d[i] = s[i];
    }
    for (int i = tid; i < GP; i += NTHR) { sB0[i] = gB0[i]; sB1[i] = gB1[i]; }
    // Zero recurrent state + padded activation regions.
    for (int i = tid; i < BT * KP; i += NTHR) { vec0[i] = (_Float16)0.f; vec1[i] = (_Float16)0.f; }
    for (int i = tid; i < BT * HH; i += NTHR) { c0[i] = 0.f; c1[i] = 0.f; h1f[i] = 0.f; }
    wait_async_all();
    __syncthreads();

    // Wave -> output-tile mapping: 2 M tiles x 20 N tiles, 5 N tiles per wave.
    const int mRow   = (wave & 1) * 16;
    const int nTile0 = (wave >> 1) * 5;

    for (int t = 0; t < TT; ++t) {
        // Convert staged x_t (f32) -> vec0[:, 0:64) (f16).
        const float* cur = xraw + (t & 1) * BT * DD;
        for (int i = tid; i < BT * DD; i += NTHR) {
            int r = i >> 6, dcol = i & 63;
            vec0[r * KP + dcol] = (_Float16)cur[r * DD + dcol];
        }
        // Prefetch x[t+1] into the other staging buffer (hidden behind GEMMs).
        if (t + 1 < TT)
            prefetch_x(x + ((size_t)(t + 1) * Bn + brow) * DD,
                       xraw + ((t + 1) & 1) * BT * DD, tid);
        __syncthreads();

        // ---- layer 0: gates = [x|h0] * [Wih0;Whh0]^T ----
        gemm_slab(vec0, sWA, gates, mRow, nTile0, half, l16);
        __syncthreads();
        for (int i = tid; i < BT * HH; i += NTHR) {
            int r = i / HH, c = i % HH;
            const float* gr = gates + (size_t)r * GP;
            float iv = sigmoidf_(gr[c]           + sB0[c]);
            float fv = sigmoidf_(gr[HH + c]      + sB0[HH + c]);
            float gv = tanhf    (gr[2 * HH + c]  + sB0[2 * HH + c]);
            float ov = sigmoidf_(gr[3 * HH + c]  + sB0[3 * HH + c]);
            float cn = fv * c0[i] + iv * gv;
            c0[i] = cn;
            float h = ov * tanhf(cn);
            _Float16 hh = (_Float16)h;
            vec0[r * KP + DD + c] = hh;   // h0 for next step's layer-0 input
            vec1[r * KP + c]      = hh;   // h0 for this step's layer-1 input
        }
        __syncthreads();

        // ---- layer 1: gates = [h0|h1] * [Wih1;Whh1]^T ----
        gemm_slab(vec1, sWB, gates, mRow, nTile0, half, l16);
        __syncthreads();
        for (int i = tid; i < BT * HH; i += NTHR) {
            int r = i / HH, c = i % HH;
            const float* gr = gates + (size_t)r * GP;
            float iv = sigmoidf_(gr[c]           + sB1[c]);
            float fv = sigmoidf_(gr[HH + c]      + sB1[HH + c]);
            float gv = tanhf    (gr[2 * HH + c]  + sB1[2 * HH + c]);
            float ov = sigmoidf_(gr[3 * HH + c]  + sB1[3 * HH + c]);
            float cn = fv * c1[i] + iv * gv;
            c1[i] = cn;
            float h = ov * tanhf(cn);
            vec1[r * KP + HH + c] = (_Float16)h;  // h1 recurrent input
            h1f[i] = h;                            // full-precision h1 for head
        }
        wait_async_all();    // next x tile resident before next conversion
        __syncthreads();
    }

    // Head: relu(h1 @ W1^T + b1) @ W2^T + b2 -> sigmoid. One row per thread.
    if (tid < BT) {
        const float* hr = h1f + (size_t)tid * HH;
        float acc2 = gb2[0];
        for (int j = 0; j < 50; ++j) {
            float a = gb1[j];
            const float* wr = gW1 + (size_t)j * HH;
            for (int k = 0; k < HH; ++k) a += wr[k] * hr[k];
            a = a > 0.f ? a : 0.f;
            acc2 += gW2[j] * a;
        }
        out[brow + tid] = sigmoidf_(acc2);
    }
}

extern "C" void kernel_launch(void* const* d_in, const int* in_sizes, int n_in,
                              void* d_out, int out_size, void* d_ws, size_t ws_size,
                              hipStream_t stream) {
    const float* data_in = (const float*)d_in[0];
    const int Bn = in_sizes[0] / (TT * DD);   // 4096
    const float* Wih0 = (const float*)d_in[2];
    const float* Whh0 = (const float*)d_in[3];
    const float* bih0 = (const float*)d_in[4];
    const float* bhh0 = (const float*)d_in[5];
    const float* Wih1 = (const float*)d_in[6];
    const float* Whh1 = (const float*)d_in[7];
    const float* bih1 = (const float*)d_in[8];
    const float* bhh1 = (const float*)d_in[9];
    const float* W1   = (const float*)d_in[10];
    const float* b1   = (const float*)d_in[11];
    const float* W2   = (const float*)d_in[12];
    const float* b2   = (const float*)d_in[13];
    float* out = (float*)d_out;

    // Workspace layout: packed f16 weights + summed biases.
    char* ws = (char*)d_ws;
    _Float16* WA = (_Float16*)ws;           // GP*KP f16
    _Float16* WB = WA + GP * KP;            // GP*KP f16
    float* B0 = (float*)(WB + GP * KP);     // GP f32
    float* B1 = B0 + GP;                    // GP f32

    pack_weights_kernel<<<(GP * KP + 255) / 256, 256, 0, stream>>>(
        Wih0, Whh0, bih0, bhh0, Wih1, Whh1, bih1, bhh1, WA, WB, B0, B1);

    const size_t smem_bytes =
        (size_t)2 * GP * KP * sizeof(_Float16) +   // sWA, sWB
        (size_t)2 * BT * KP * sizeof(_Float16) +   // vec0, vec1
        (size_t)BT * GP * sizeof(float) +          // gates
        (size_t)3 * BT * HH * sizeof(float) +      // c0, c1, h1f
        (size_t)2 * GP * sizeof(float) +           // sB0, sB1
        (size_t)2 * BT * DD * sizeof(float);       // xraw double buffer

    lstm_seq_kernel<<<Bn / BT, NTHR, smem_bytes, stream>>>(
        data_in, WA, WB, B0, B1, W1, b1, W2, b2, out, Bn);
}
